// Block_28930899705927
// MI455X (gfx1250) — matrix operand order repaired and checked by reading, not compiled, not run on hardware
//
#include <hip/hip_runtime.h>
#include <hip/hip_bf16.h>

typedef _Float16 v16h __attribute__((ext_vector_type(16)));
typedef _Float16 v8h  __attribute__((ext_vector_type(8)));
typedef float    v8f  __attribute__((ext_vector_type(8)));
typedef unsigned int u32x4 __attribute__((ext_vector_type(4)));
typedef int          i32x4 __attribute__((ext_vector_type(4)));
typedef int          i32x8 __attribute__((ext_vector_type(8)));

union Frag { v16h v; v8h h[2]; };

// ---- LDS layout (bytes) ----
constexpr int XS0_OFF  = 0;        // float[64*64] x tile, buffer 0
constexpr int XS1_OFF  = 16384;    // float[64*64] x tile, buffer 1 (TDM double buffer)
constexpr int LN1W_OFF = 32768;    // float[64]
constexpr int LN1B_OFF = 33024;    // float[64]
constexpr int LN2W_OFF = 33280;    // float[64]
constexpr int LN2B_OFF = 33536;    // float[64]
constexpr int PJB_OFF  = 33792;    // float[64]
constexpr int F2B_OFF  = 34048;    // float[64]
constexpr int F1B_OFF  = 34304;    // float[256] -> ends 35328
constexpr int WQT_OFF  = 35328;    // f16[64][64]  Wq^T
constexpr int WKT_OFF  = 43520;    // f16[64][64]  Wk^T
constexpr int WVT_OFF  = 51712;    // f16[64][64]  Wv^T
constexpr int PWT_OFF  = 59904;    // f16[64][64]  proj_w^T
constexpr int F1T_OFF  = 68096;    // f16[256][64] fc1_w^T
constexpr int F2T_OFF  = 100864;   // f16[64][256] fc2_w^T
constexpr int HS_OFF   = 133632;   // f16[64][64]  LN output (A operand)
constexpr int QB_OFF   = 141824;   // f16[64][64]  q (pre-scaled by 1/8)
constexpr int KB_OFF   = 150016;   // f16[64][64]  k (row-major == B^T for scores)
constexpr int VT_OFF   = 158208;   // f16[64][64]  v transposed [c][s]
constexpr int AT_OFF   = 166400;   // f16[64][64]  attn concat
constexpr int FF_OFF   = 174592;   // f16[64][256] relu(fc1)
constexpr int SC_OFF   = 207360;   // f16[8][64][64] scores; aliased float[64*64] obuf
constexpr int SMEM_TOTAL = 272896; // 266.5 KB < 320 KB / WGP

__device__ __forceinline__ v8f wmma_f16(v16h a, v16h b, v8f c) {
  return __builtin_amdgcn_wmma_f32_16x16x32_f16(false, a, false, b, (short)0, c,
                                                false, false);
}

// ---- TDM descriptor builders (CDNA5 ISA) ----
// Group0: count=1 (valid user D#), lds_addr, 57-bit global addr, type=2 ("image")
__device__ __forceinline__ u32x4 tdm_g0(unsigned lds_off, unsigned long long ga) {
  u32x4 g;
  g[0] = 1u;                                          // count=1
  g[1] = lds_off;                                     // lds_addr [63:32]
  g[2] = (unsigned)(ga & 0xFFFFFFFFull);              // global_addr [95:64]
  g[3] = (unsigned)((ga >> 32) & 0x1FFFFFFull) | (2u << 30); // addr hi | type=2
  return g;
}
// Group1: 1-D tile of `n` 4-byte elements (n <= 65535): data_size=4B,
// tensor_dim0=n, tensor_dim1=1, tile_dim0=n, tile_dim1=1, dim0_stride=n.
__device__ __forceinline__ i32x8 tdm_g1_1d(unsigned n) {
  i32x8 g;
  g[0] = (int)(2u << 16);          // data_size=2 (4 bytes); mask/flags = 0
  g[1] = (int)(n << 16);           // tensor_dim0[15:0] @ bits 63:48
  g[2] = (int)(1u << 16);          // tensor_dim0 hi=0; tensor_dim1[15:0]=1
  g[3] = (int)(n << 16);           // tensor_dim1 hi=0; tile_dim0=n
  g[4] = 1;                        // tile_dim1=1, tile_dim2=0
  g[5] = (int)n;                   // tensor_dim0_stride[31:0]
  g[6] = 0;                        // stride hi / dim1_stride
  g[7] = 0;
  return g;
}

__device__ __forceinline__ void tdm_load(u32x4 g0, i32x8 g1) {
  i32x4 z4 = {};
#if __clang_major__ >= 23
  i32x8 z8 = {};
  __builtin_amdgcn_tensor_load_to_lds(g0, g1, z4, z4, z8, 0);
#else
  __builtin_amdgcn_tensor_load_to_lds(g0, g1, z4, z4, 0);
#endif
}
__device__ __forceinline__ void tdm_store(u32x4 g0, i32x8 g1) {
  i32x4 z4 = {};
#if __clang_major__ >= 23
  i32x8 z8 = {};
  __builtin_amdgcn_tensor_store_from_lds(g0, g1, z4, z4, z8, 0);
#else
  __builtin_amdgcn_tensor_store_from_lds(g0, g1, z4, z4, 0);
#endif
}

// A fragment 16x32 f16 from row-major LDS [*, ld]; rows m0..m0+15, K kb..kb+31.
__device__ __forceinline__ v16h load_a(const _Float16* base, int ld, int m0,
                                       int kb, int lane) {
  int r  = m0 + (lane & 15);
  int ko = kb + ((lane & 16) ? 16 : 0);
  const _Float16* p = base + r * ld + ko;
  Frag u;
  u.h[0] = *(const v8h*)p;
  u.h[1] = *(const v8h*)(p + 8);
  return u.v;
}

// B fragment 32x16 f16 from B^T row-major [N][ldk]; cols n0..n0+15, K kb..kb+31.
__device__ __forceinline__ v16h load_bt(const _Float16* baseT, int ldk, int n0,
                                        int kb, int lane) {
  int n  = n0 + (lane & 15);
  int ko = kb + ((lane & 16) ? 16 : 0);
  const _Float16* p = baseT + n * ldk + ko;
  Frag u;
  u.h[0] = *(const v8h*)p;
  u.h[1] = *(const v8h*)(p + 8);
  return u.v;
}

// A fragment with only K=0..7 valid (head-dim 8, zero-padded to 32).
__device__ __forceinline__ v16h load_a_k8(const _Float16* base, int ld, int m0,
                                          int d0, int lane) {
  Frag u = {};
  if (lane < 16) u.h[0] = *(const v8h*)(base + (m0 + lane) * ld + d0);
  return u.v;
}

// B fragment (K=0..7 valid) where B^T is row-major kbuf [s][ld].
__device__ __forceinline__ v16h load_bt_k8(const _Float16* baseT, int ld, int n0,
                                           int d0, int lane) {
  Frag u = {};
  if (lane < 16) u.h[0] = *(const v8h*)(baseT + (n0 + lane) * ld + d0);
  return u.v;
}

__device__ __forceinline__ void store_d_f16(_Float16* dst, int ld, int m0, int n0,
                                            int lane, v8f acc, float scale) {
  int rb = m0 + ((lane & 16) ? 8 : 0);
  int c  = n0 + (lane & 15);
#pragma unroll
  for (int r = 0; r < 8; ++r) dst[(rb + r) * ld + c] = (_Float16)(acc[r] * scale);
}

__device__ __forceinline__ void layernorm_ph(const float* xs, _Float16* hs,
                                             const float* w, const float* b,
                                             int tid) {
  int row = tid >> 3;
  int c0  = (tid & 7) << 3;
  const float* xr = xs + row * 64 + c0;
  float xv[8];
  float s = 0.f;
#pragma unroll
  for (int i = 0; i < 8; ++i) { xv[i] = xr[i]; s += xv[i]; }
#pragma unroll
  for (int m = 1; m < 8; m <<= 1) s += __shfl_xor(s, m, 8);
  float mu = s * (1.0f / 64.0f);
  float vs = 0.f;
#pragma unroll
  for (int i = 0; i < 8; ++i) { float d = xv[i] - mu; vs += d * d; }
#pragma unroll
  for (int m = 1; m < 8; m <<= 1) vs += __shfl_xor(vs, m, 8);
  float rstd = rsqrtf(vs * (1.0f / 64.0f) + 1e-5f);
  _Float16* hr = hs + row * 64 + c0;
#pragma unroll
  for (int i = 0; i < 8; ++i)
    hr[i] = (_Float16)((xv[i] - mu) * rstd * w[c0 + i] + b[c0 + i]);
}

__global__ void __launch_bounds__(512, 1)
gpt_block_kernel(const float* __restrict__ xg,
                 const float* __restrict__ g_ln1w, const float* __restrict__ g_ln1b,
                 const float* __restrict__ g_wq, const float* __restrict__ g_wk,
                 const float* __restrict__ g_wv,
                 const float* __restrict__ g_pw, const float* __restrict__ g_pb,
                 const float* __restrict__ g_ln2w, const float* __restrict__ g_ln2b,
                 const float* __restrict__ g_f1, const float* __restrict__ g_f1b,
                 const float* __restrict__ g_f2, const float* __restrict__ g_f2b,
                 float* __restrict__ outg, int nbatch) {
  extern __shared__ char smem[];
  float* xs0     = (float*)(smem + XS0_OFF);
  float* xs1     = (float*)(smem + XS1_OFF);
  float* s_ln1w  = (float*)(smem + LN1W_OFF);
  float* s_ln1b  = (float*)(smem + LN1B_OFF);
  float* s_ln2w  = (float*)(smem + LN2W_OFF);
  float* s_ln2b  = (float*)(smem + LN2B_OFF);
  float* s_pb    = (float*)(smem + PJB_OFF);
  float* s_f2b   = (float*)(smem + F2B_OFF);
  float* s_f1b   = (float*)(smem + F1B_OFF);
  _Float16* wqT  = (_Float16*)(smem + WQT_OFF);
  _Float16* wkT  = (_Float16*)(smem + WKT_OFF);
  _Float16* wvT  = (_Float16*)(smem + WVT_OFF);
  _Float16* pwT  = (_Float16*)(smem + PWT_OFF);
  _Float16* f1T  = (_Float16*)(smem + F1T_OFF);
  _Float16* f2T  = (_Float16*)(smem + F2T_OFF);
  _Float16* hs   = (_Float16*)(smem + HS_OFF);
  _Float16* qb   = (_Float16*)(smem + QB_OFF);
  _Float16* kbuf = (_Float16*)(smem + KB_OFF);
  _Float16* vT   = (_Float16*)(smem + VT_OFF);
  _Float16* attn = (_Float16*)(smem + AT_OFF);
  _Float16* ff1  = (_Float16*)(smem + FF_OFF);
  _Float16* sc   = (_Float16*)(smem + SC_OFF);
  float* obuf    = (float*)(smem + SC_OFF);   // alias: scores dead by fc2 time

  const int tid = threadIdx.x, lane = tid & 31, wid = tid >> 5;
  const int b0 = blockIdx.x * 4;
  const int nb_local = (nbatch - b0 < 4) ? (nbatch - b0) : 4;
  if (nb_local <= 0) return;

  // ---- Kick off async TDM load of the first x tile (overlaps weight staging) ----
  if (wid == 0) {
    tdm_load(tdm_g0((unsigned)XS0_OFF,
                    (unsigned long long)(xg + (size_t)b0 * 4096)),
             tdm_g1_1d(4096u));
  }

  // ---- Stage all weights into LDS as f16, B-transposed (once per workgroup) ----
  for (int i = tid; i < 4096; i += 512) {
    int c = i >> 6, j = i & 63;
    int src = ((j >> 3) << 9) + (c << 3) + (j & 7);  // [H][C][hs] -> col j=h*8+d
    wqT[j * 64 + c] = (_Float16)g_wq[src];
    wkT[j * 64 + c] = (_Float16)g_wk[src];
    wvT[j * 64 + c] = (_Float16)g_wv[src];
    pwT[j * 64 + c] = (_Float16)g_pw[c * 64 + j];
  }
  for (int i = tid; i < 16384; i += 512) {
    int c1 = i >> 8, j1 = i & 255;          // fc1_w [64][256]
    f1T[j1 * 64 + c1] = (_Float16)g_f1[c1 * 256 + j1];
    int c2 = i >> 6, j2 = i & 63;           // fc2_w [256][64]
    f2T[j2 * 256 + c2] = (_Float16)g_f2[c2 * 64 + j2];
  }
  if (tid < 64) {
    s_ln1w[tid] = g_ln1w[tid]; s_ln1b[tid] = g_ln1b[tid];
    s_ln2w[tid] = g_ln2w[tid]; s_ln2b[tid] = g_ln2b[tid];
    s_pb[tid] = g_pb[tid];     s_f2b[tid] = g_f2b[tid];
  }
  if (tid < 256) s_f1b[tid] = g_f1b[tid];

  // ---- Pipelined batch loop: TDM prefetch next tile while computing current ----
  for (int bi = 0; bi < nb_local; ++bi) {
    int b = b0 + bi;
    float* xs = (bi & 1) ? xs1 : xs0;
    bool has_next = (bi + 1) < nb_local;

    if (wid == 0) {
      if (has_next) {
        unsigned lds_next = (unsigned)(((bi + 1) & 1) ? XS1_OFF : XS0_OFF);
        tdm_load(tdm_g0(lds_next,
                        (unsigned long long)(xg + (size_t)(b + 1) * 4096)),
                 tdm_g1_1d(4096u));
        // In-order per-wave tensor ops: tcnt<=1 => load(bi) AND store(bi-1) done.
        __builtin_amdgcn_s_wait_tensorcnt(1);
      } else {
        __builtin_amdgcn_s_wait_tensorcnt(0);
      }
    }
    __syncthreads();   // x tile in LDS + obuf/sc region free, visible to all waves

    layernorm_ph(xs, hs, s_ln1w, s_ln1b, tid);
    __syncthreads();

    // ---- QKV: three 64x64x64 GEMMs; each wave owns one 16x16 tile of each ----
    {
      int tm = (wid >> 2) * 16, tn = (wid & 3) * 16;
      v16h a0 = load_a(hs, 64, tm, 0, lane);
      v16h a1 = load_a(hs, 64, tm, 32, lane);

      v8f accq = {};
      accq = wmma_f16(a0, load_bt(wqT, 64, tn, 0, lane), accq);
      accq = wmma_f16(a1, load_bt(wqT, 64, tn, 32, lane), accq);
      store_d_f16(qb, 64, tm, tn, lane, accq, 0.125f);  // fold C^-0.5

      v8f acck = {};
      acck = wmma_f16(a0, load_bt(wkT, 64, tn, 0, lane), acck);
      acck = wmma_f16(a1, load_bt(wkT, 64, tn, 32, lane), acck);
      store_d_f16(kbuf, 64, tm, tn, lane, acck, 1.0f);

      v8f accv = {};
      accv = wmma_f16(a0, load_bt(wvT, 64, tn, 0, lane), accv);
      accv = wmma_f16(a1, load_bt(wvT, 64, tn, 32, lane), accv);
      int rb = tm + ((lane & 16) ? 8 : 0);
      int c  = tn + (lane & 15);
      v8h pk;                                  // transposed store vT[c][s],
#pragma unroll                                 // contiguous in r -> 1x b128
      for (int r = 0; r < 8; ++r) pk[r] = (_Float16)accv[r];
      *(v8h*)(vT + c * 64 + rb) = pk;
    }
    __syncthreads();

    // ---- scores = q k^T (per head, K=8 zero-padded; lower/diag tiles only) ----
    {
      const int TMt[10] = {0, 1, 1, 2, 2, 2, 3, 3, 3, 3};
      const int TNt[10] = {0, 0, 1, 0, 1, 2, 0, 1, 2, 3};
#pragma unroll
      for (int it = 0; it < 5; ++it) {
        int g = wid + 16 * it;             // 0..79
        int h = g / 10, p = g % 10;
        v16h a  = load_a_k8(qb, 64, TMt[p] * 16, h * 8, lane);
        v16h bq = load_bt_k8(kbuf, 64, TNt[p] * 16, h * 8, lane);
        v8f acc = {};
        acc = wmma_f16(a, bq, acc);
        store_d_f16(sc + h * 4096, 64, TMt[p] * 16, TNt[p] * 16, lane, acc, 1.0f);
      }
    }
    __syncthreads();

    // ---- causal softmax: one row per thread (8 heads x 64 rows = 512) ----
    {
      int h = tid >> 6, t = tid & 63;
      _Float16* row = sc + h * 4096 + t * 64;
      float mx = -3.0e38f;
      for (int s2 = 0; s2 <= t; ++s2) mx = fmaxf(mx, (float)row[s2]);
      float sum = 0.f;
      for (int s2 = 0; s2 <= t; ++s2) {
        float e = __expf((float)row[s2] - mx);
        sum += e;
        row[s2] = (_Float16)e;
      }
      float inv = 1.0f / sum;
      for (int s2 = 0; s2 <= t; ++s2)
        row[s2] = (_Float16)((float)row[s2] * inv);
      for (int s2 = t + 1; s2 < 64; ++s2) row[s2] = (_Float16)0.0f;
    }
    __syncthreads();

    // ---- attn = softmax(scores) @ v (per head, N=8 zero-padded) ----
    {
#pragma unroll
      for (int it = 0; it < 2; ++it) {
        int g = wid + 16 * it;             // 0..31: head = g/4, tm = g%4
        int h = g >> 2, tm = (g & 3) * 16;
        v8f acc = {};
#pragma unroll
        for (int kk = 0; kk < 64; kk += 32) {
          v16h a = load_a(sc + h * 4096, 64, tm, kk, lane);
          Frag u = {};
          int n = lane & 15;
          if (n < 8) {                     // B(k,n)=v[k][h*8+n] -> vT rows
            const _Float16* p = vT + (h * 8 + n) * 64 + kk + ((lane & 16) ? 16 : 0);
            u.h[0] = *(const v8h*)p;
            u.h[1] = *(const v8h*)(p + 8);
          }
          acc = wmma_f16(a, u.v, acc);
        }
        int n = lane & 15;
        if (n < 8) {
          int rb = tm + ((lane & 16) ? 8 : 0);
#pragma unroll
          for (int r = 0; r < 8; ++r)
            attn[(rb + r) * 64 + h * 8 + n] = (_Float16)acc[r];
        }
      }
    }
    __syncthreads();

    // ---- proj + bias + residual -> xs (f32, in place) ----
    {
      int tm = (wid >> 2) * 16, tn = (wid & 3) * 16;
      v8f acc = {};
      acc = wmma_f16(load_a(attn, 64, tm, 0, lane),  load_bt(pwT, 64, tn, 0, lane),  acc);
      acc = wmma_f16(load_a(attn, 64, tm, 32, lane), load_bt(pwT, 64, tn, 32, lane), acc);
      int rb = tm + ((lane & 16) ? 8 : 0);
      int c  = tn + (lane & 15);
#pragma unroll
      for (int r = 0; r < 8; ++r) {
        int idx = (rb + r) * 64 + c;
        xs[idx] = xs[idx] + acc[r] + s_pb[c];
      }
    }
    __syncthreads();

    layernorm_ph(xs, hs, s_ln2w, s_ln2b, tid);
    __syncthreads();

    // ---- fc1 + bias + relu -> ff1 f16 [64][256] ----
    {
#pragma unroll
      for (int it = 0; it < 4; ++it) {
        int g = wid + 16 * it;             // 0..63: tm = g/16, tn = g%16
        int tm = (g >> 4) * 16, tn = (g & 15) * 16;
        v8f acc = {};
        acc = wmma_f16(load_a(hs, 64, tm, 0, lane),  load_bt(f1T, 64, tn, 0, lane),  acc);
        acc = wmma_f16(load_a(hs, 64, tm, 32, lane), load_bt(f1T, 64, tn, 32, lane), acc);
        int rb = tm + ((lane & 16) ? 8 : 0);
        int c  = tn + (lane & 15);
#pragma unroll
        for (int r = 0; r < 8; ++r)
          ff1[(rb + r) * 256 + c] = (_Float16)fmaxf(acc[r] + s_f1b[c], 0.0f);
      }
    }
    __syncthreads();

    // ---- fc2 + bias + residual -> obuf f32 ----
    {
      int tm = (wid >> 2) * 16, tn = (wid & 3) * 16;
      v8f acc = {};
#pragma unroll
      for (int kk = 0; kk < 256; kk += 32)
        acc = wmma_f16(load_a(ff1, 256, tm, kk, lane),
                       load_bt(f2T, 256, tn, kk, lane), acc);
      int rb = tm + ((lane & 16) ? 8 : 0);
      int c  = tn + (lane & 15);
#pragma unroll
      for (int r = 0; r < 8; ++r) {
        int idx = (rb + r) * 64 + c;
        obuf[idx] = xs[idx] + acc[r] + s_f2b[c];
      }
    }
    __syncthreads();   // obuf complete (DScnt drained) before TDM store reads it

    // ---- async TDM store of the finished tile (LDS -> HBM) ----
    if (wid == 0) {
      tdm_store(tdm_g0((unsigned)SC_OFF,
                       (unsigned long long)(outg + (size_t)b * 4096)),
                tdm_g1_1d(4096u));
    }
    // Completion enforced by next iteration's s_wait_tensorcnt (obuf aliases sc)
    // or by the s_wait_tensorcnt(0) after the loop on the last iteration.
  }

  if (wid == 0) __builtin_amdgcn_s_wait_tensorcnt(0);
}

extern "C" void kernel_launch(void* const* d_in, const int* in_sizes, int n_in,
                              void* d_out, int out_size, void* d_ws, size_t ws_size,
                              hipStream_t stream) {
  (void)n_in; (void)out_size; (void)d_ws; (void)ws_size;
  const float* x     = (const float*)d_in[0];
  const float* ln1w  = (const float*)d_in[1];
  const float* ln1b  = (const float*)d_in[2];
  const float* wq    = (const float*)d_in[3];
  const float* wk    = (const float*)d_in[4];
  const float* wv    = (const float*)d_in[5];
  const float* pw    = (const float*)d_in[6];
  const float* pb    = (const float*)d_in[7];
  const float* ln2w  = (const float*)d_in[8];
  const float* ln2b  = (const float*)d_in[9];
  const float* f1    = (const float*)d_in[10];
  const float* f1b   = (const float*)d_in[11];
  const float* f2    = (const float*)d_in[12];
  const float* f2b   = (const float*)d_in[13];
  float* out = (float*)d_out;

  int nbatch = in_sizes[0] / (64 * 64);
  int grid = (nbatch + 3) / 4;

  (void)hipFuncSetAttribute(reinterpret_cast<const void*>(gpt_block_kernel),
                            hipFuncAttributeMaxDynamicSharedMemorySize, SMEM_TOTAL);
  gpt_block_kernel<<<grid, 512, SMEM_TOTAL, stream>>>(
      x, ln1w, ln1b, wq, wk, wv, pw, pb, ln2w, ln2b, f1, f1b, f2, f2b,
      out, nbatch);
}
